// UTDGraphNetNoise_6176162972384
// MI455X (gfx1250) — matrix operand order repaired and checked by reading, not compiled, not run on hardware
//
#include <hip/hip_runtime.h>
#include <hip/hip_bf16.h>
#include <math.h>

#define N_NODES 100000
#define HIDDEN  128
#define TAU_TH  0.005f

typedef __attribute__((ext_vector_type(16))) __bf16 v16bf;
typedef __attribute__((ext_vector_type(8)))  __bf16 v8bf;
typedef __attribute__((ext_vector_type(4)))  __bf16 v4bf;
typedef __attribute__((ext_vector_type(8)))  float  v8f;

static __device__ __forceinline__ __bf16 f2bf(float f) {
    union { float f; unsigned u; } c; c.f = f;
    unsigned short hb = (unsigned short)(c.u >> 16);
    __bf16 r; __builtin_memcpy(&r, &hb, 2); return r;
}

// ---------------------------------------------------------------------------
// f32 -> bf16 streaming convert (float4 in, 8B bf16x4 out)
// ---------------------------------------------------------------------------
__global__ __launch_bounds__(256) void cvt_bf16_kernel(
    const float* __restrict__ s, __bf16* __restrict__ d, long n)
{
    long n4 = n >> 2;
    const float4* s4 = (const float4*)s;
    v4bf* d4 = (v4bf*)d;
    for (long i = (long)blockIdx.x * blockDim.x + threadIdx.x; i < n4;
         i += (long)gridDim.x * blockDim.x) {
        float4 v = s4[i];
        v4bf o;
        o[0] = f2bf(v.x); o[1] = f2bf(v.y); o[2] = f2bf(v.z); o[3] = f2bf(v.w);
        d4[i] = o;
    }
}

// ---------------------------------------------------------------------------
// WMMA GEMM, bf16 operands, f32 accum:
//   C[M x Nout] = act( [A1 | A2][M x (K1+K2)] @ W^T + bias )
// One block = one 16-row M tile; 8 waves = 8 N tiles. A tile staged in LDS
// (<=8KB), fragments read via ds_load_b128. W rows read from global (L2-hot)
// with the row index CLAMPED (not predicated): WMMA column n depends only on
// B column n, and columns >= Nout are never stored, so clamped-garbage is
// harmless and the inner loop stays branch-free.
// Ktot must be 128 or 256 (power-of-two splits below).
// ---------------------------------------------------------------------------
__global__ __launch_bounds__(256) void wmma_gemm_kernel(
    const __bf16* __restrict__ A1, const __bf16* __restrict__ A2,
    const __bf16* __restrict__ W,  const float* __restrict__ bias,
    float* __restrict__ C,
    int K1, int K2, int Nout, int ldc, int act)
{
    __shared__ __bf16 s_a[16 * 256];
    const int Ktot = K1 + K2;
    const int m0   = blockIdx.x * 16;

    // cooperative LDS fill: 8B (4 x bf16) chunks, row-major [16][Ktot]
    const int nch = (16 * Ktot) >> 2;
    for (int ch = threadIdx.x; ch < nch; ch += 256) {
        int k4  = ch << 2;
        int row = k4 / Ktot;          // Ktot is 128/256 -> shift
        int k   = k4 - row * Ktot;
        const __bf16* src = (k < K1)
            ? A1 + (size_t)(m0 + row) * K1 + k
            : A2 + (size_t)(m0 + row) * K2 + (k - K1);
        v4bf t; __builtin_memcpy(&t, src, 8);
        *(v4bf*)(s_a + row * Ktot + k) = t;
    }
    __syncthreads();

    const int wv   = threadIdx.x >> 5;
    const int lane = threadIdx.x & 31;
    const int half = lane >> 4;
    const int l16  = lane & 15;
    const int n0   = wv * 16;
    if (n0 >= Nout) return;          // inactive N tiles (after barrier)
    const int ncol   = n0 + l16;
    const bool colok = (ncol < Nout);
    const int ncl    = colok ? ncol : (Nout - 1);   // clamped, always valid

    float bv = bias[ncl];
    v8f acc;
#pragma unroll
    for (int r = 0; r < 8; ++r) acc[r] = bv;

    const __bf16* arow = s_a + l16 * Ktot;
    const __bf16* wrow = W + (size_t)ncl * Ktot;

    for (int kt = 0; kt < Ktot; kt += 32) {
        v16bf afr, bfr;
        // A fragment (ISA 16x32 bf16 layout): lane<16 K={0..7,16..23}
        __builtin_memcpy(&afr,             arow + kt + half * 8,      16);
        __builtin_memcpy((char*)&afr + 16, arow + kt + half * 8 + 16, 16);
        // B fragment (same layout as A): lane<16 K={0..7,16..23}
        __builtin_memcpy(&bfr,             wrow + kt + half * 8,      16);
        __builtin_memcpy((char*)&bfr + 16, wrow + kt + half * 8 + 16, 16);
        acc = __builtin_amdgcn_wmma_f32_16x16x32_bf16(false, afr, false, bfr,
                                                      (short)0, acc, false, false);
    }

    if (colok) {
#pragma unroll
        for (int r = 0; r < 8; ++r) {
            int row = m0 + r + half * 8;
            float v = acc[r];
            if (act) v = fmaxf(v, 0.0f);
            C[(size_t)row * ldc + ncol] = v;
        }
    }
}

// ---------------------------------------------------------------------------
// Zero-fill (float4 grid-stride)
// ---------------------------------------------------------------------------
__global__ __launch_bounds__(256) void zero_kernel(float* __restrict__ p, long n)
{
    float4* p4 = (float4*)p;
    long n4 = n >> 2;
    float4 z; z.x = z.y = z.z = z.w = 0.0f;
    for (long i = (long)blockIdx.x * blockDim.x + threadIdx.x; i < n4;
         i += (long)gridDim.x * blockDim.x)
        p4[i] = z;
}

// ---------------------------------------------------------------------------
// Edge phase: one wave per edge; float4 gathers from L2-resident h, abs-diff,
// global_atomic_add_f32 scatter into agg[row]; global_prefetch_b8 next rows.
// ---------------------------------------------------------------------------
__global__ __launch_bounds__(256) void edge_kernel(
    const float* __restrict__ h, const int* __restrict__ ei,
    float* __restrict__ agg, int E)
{
    const int lane   = threadIdx.x & 31;
    const int waveId = (blockIdx.x * blockDim.x + threadIdx.x) >> 5;
    const int nWaves = (gridDim.x * blockDim.x) >> 5;

    for (int e = waveId; e < E; e += nWaves) {
        int r = ei[e];
        int c = ei[E + e];
        int e2 = e + nWaves;
        if (e2 < E) {
            int r2 = ei[e2], c2 = ei[E + e2];
            __builtin_prefetch(h + (size_t)r2 * HIDDEN, 0, 0);
            __builtin_prefetch(h + (size_t)c2 * HIDDEN, 0, 0);
        }
        const float4* hr = (const float4*)(h + (size_t)r * HIDDEN);
        const float4* hc = (const float4*)(h + (size_t)c * HIDDEN);
        float4 a = hr[lane];
        float4 b = hc[lane];
        float4 d;
        d.x = fabsf(a.x - b.x); d.y = fabsf(a.y - b.y);
        d.z = fabsf(a.z - b.z); d.w = fabsf(a.w - b.w);
        float* dst = agg + (size_t)r * HIDDEN + lane * 4;
        atomicAdd(dst + 0, d.x);
        atomicAdd(dst + 1, d.y);
        atomicAdd(dst + 2, d.z);
        atomicAdd(dst + 3, d.w);
    }
}

// ---------------------------------------------------------------------------
// Fused GRU + tau + mask. One block = one 16-row tile, 8 waves.
// bf16 h/agg tiles staged in LDS (8KB); 48 WMMA tiles of gi/gh into 48KB LDS;
// then tau = softplus(h.Wt+bt) and the masked gated update of f32 h in place.
// ---------------------------------------------------------------------------
__global__ __launch_bounds__(256) void gru_kernel(
    float* __restrict__ h,
    const __bf16* __restrict__ hbf,  const __bf16* __restrict__ aggbf,
    const __bf16* __restrict__ Wih,  const __bf16* __restrict__ Whh,
    const float* __restrict__ b_ih,  const float* __restrict__ b_hh,
    const float* __restrict__ Wt,    const float* __restrict__ bt,
    float* __restrict__ tau_out)
{
    __shared__ float  s_gi[16 * 384];
    __shared__ float  s_gh[16 * 384];
    __shared__ __bf16 s_ha[16 * HIDDEN];
    __shared__ __bf16 s_ag[16 * HIDDEN];
    __shared__ float  s_tau[16];

    const int m0   = blockIdx.x * 16;
    const int wv   = threadIdx.x >> 5;
    const int lane = threadIdx.x & 31;
    const int half = lane >> 4;
    const int l16  = lane & 15;

    // Stage bf16 A tiles (16x128 each) into LDS, 8B chunks.
    for (int ch = threadIdx.x; ch < (16 * HIDDEN) >> 2; ch += 256) {
        int off = ch << 2;
        int row = off >> 7, k = off & (HIDDEN - 1);
        size_t g = (size_t)(m0 + row) * HIDDEN + k;
        v4bf t0; __builtin_memcpy(&t0, hbf  + g, 8); *(v4bf*)(s_ha + off) = t0;
        v4bf t1; __builtin_memcpy(&t1, aggbf + g, 8); *(v4bf*)(s_ag + off) = t1;
    }
    __syncthreads();

    // Phase 1: 48 WMMA tiles (24 gi: agg@Wih^T, 24 gh: h@Whh^T), 6 per wave.
    for (int t = wv; t < 48; t += 8) {
        const bool   is_gh = (t >= 24);
        const int    n0    = (t % 24) * 16;
        const __bf16* A    = is_gh ? s_ha : s_ag;
        const __bf16* W    = is_gh ? Whh  : Wih;
        const float*  bias = is_gh ? b_hh : b_ih;

        float bv = bias[n0 + l16];
        v8f acc;
#pragma unroll
        for (int r = 0; r < 8; ++r) acc[r] = bv;

        const __bf16* arow = A + l16 * HIDDEN;
        const __bf16* wrow = W + (size_t)(n0 + l16) * HIDDEN;
#pragma unroll
        for (int kt = 0; kt < HIDDEN; kt += 32) {
            v16bf afr, bfr;
            __builtin_memcpy(&afr,             arow + kt + half * 8,      16);
            __builtin_memcpy((char*)&afr + 16, arow + kt + half * 8 + 16, 16);
            __builtin_memcpy(&bfr,             wrow + kt + half * 8,      16);
            __builtin_memcpy((char*)&bfr + 16, wrow + kt + half * 8 + 16, 16);
            acc = __builtin_amdgcn_wmma_f32_16x16x32_bf16(false, afr, false, bfr,
                                                          (short)0, acc, false, false);
        }
        float* dst = is_gh ? s_gh : s_gi;
#pragma unroll
        for (int r = 0; r < 8; ++r)
            dst[(r + half * 8) * 384 + n0 + l16] = acc[r];
    }
    __syncthreads();

    // Phase 2: tau = softplus(h . Wt + bt) per row.
    if (threadIdx.x < 16) {
        const float* hrow = h + (size_t)(m0 + threadIdx.x) * HIDDEN;
        float d = bt[0];
        for (int k = 0; k < HIDDEN; ++k) d += hrow[k] * Wt[k];
        float tau = (d > 20.0f) ? d : log1pf(expf(d));
        s_tau[threadIdx.x] = tau;
        tau_out[m0 + threadIdx.x] = tau;
    }
    __syncthreads();

    // Phase 3: gates + masked update (repeated GRU on same (x,h) is idempotent).
    for (int idx = threadIdx.x; idx < 16 * HIDDEN; idx += 256) {
        int row = idx >> 7;
        int c   = idx & (HIDDEN - 1);
        float ir  = s_gi[row * 384 + c];
        float iz  = s_gi[row * 384 + 128 + c];
        float in_ = s_gi[row * 384 + 256 + c];
        float hr  = s_gh[row * 384 + c];
        float hz  = s_gh[row * 384 + 128 + c];
        float hn  = s_gh[row * 384 + 256 + c];
        size_t gidx = (size_t)(m0 + row) * HIDDEN + c;
        float hcur = h[gidx];
        float rg = 1.0f / (1.0f + expf(-(ir + hr)));
        float zg = 1.0f / (1.0f + expf(-(iz + hz)));
        float ng = tanhf(in_ + rg * hn);
        float hg = (1.0f - zg) * ng + zg * hcur;
        bool mask = (s_tau[row] < TAU_TH);
        h[gidx] = mask ? hg : hcur;
    }
}

// ---------------------------------------------------------------------------
extern "C" void kernel_launch(void* const* d_in, const int* in_sizes, int n_in,
                              void* d_out, int out_size, void* d_ws, size_t ws_size,
                              hipStream_t stream)
{
    (void)n_in; (void)out_size; (void)ws_size;
    const float* x    = (const float*)d_in[0];
    const int*   ei   = (const int*)  d_in[1];
    const float* W_in = (const float*)d_in[2];
    const float* b_in = (const float*)d_in[3];
    const float* Wd   = (const float*)d_in[4];
    const float* bd   = (const float*)d_in[5];
    const float* Wt   = (const float*)d_in[6];
    const float* bt   = (const float*)d_in[7];
    const float* W_ih = (const float*)d_in[8];
    const float* W_hh = (const float*)d_in[9];
    const float* b_ih = (const float*)d_in[10];
    const float* b_hh = (const float*)d_in[11];
    const float* Wo   = (const float*)d_in[12];
    const float* bo   = (const float*)d_in[13];
    const int E = in_sizes[1] / 2;

    const size_t NH = (size_t)N_NODES * HIDDEN;

    // f32 state
    float* hA  = (float*)d_ws;
    float* hB  = hA + NH;
    float* agg = hB + NH;
    // bf16 mirrors + weights
    __bf16* xbf    = (__bf16*)(agg + NH);
    __bf16* hbf    = xbf  + NH;
    __bf16* aggbf  = hbf  + NH;
    __bf16* Winbf  = aggbf + NH;                 // 128*128
    __bf16* Wdbf   = Winbf + 128 * 128;          // 2*128*256
    __bf16* Wihbf  = Wdbf  + 2 * 128 * 256;      // 384*128
    __bf16* Whhbf  = Wihbf + 384 * 128;          // 384*128
    __bf16* Wobf   = Whhbf + 384 * 128;          // 2*128

    float* out     = (float*)d_out;              // [N,2]
    float* tau_out = out + (size_t)N_NODES * 2;  // [N]

    const int MT = N_NODES / 16;                 // 6250 exact
    dim3 b256(256);

    // one-time bf16 conversions (deterministic every call)
    cvt_bf16_kernel<<<4096, b256, 0, stream>>>(x,    xbf,   (long)NH);
    cvt_bf16_kernel<<<64,   b256, 0, stream>>>(W_in, Winbf, 128 * 128);
    cvt_bf16_kernel<<<64,   b256, 0, stream>>>(Wd,   Wdbf,  2 * 128 * 256);
    cvt_bf16_kernel<<<64,   b256, 0, stream>>>(W_ih, Wihbf, 384 * 128);
    cvt_bf16_kernel<<<64,   b256, 0, stream>>>(W_hh, Whhbf, 384 * 128);
    cvt_bf16_kernel<<<1,    b256, 0, stream>>>(Wo,   Wobf,  2 * 128);

    // h = relu(x @ W_in^T + b_in)
    wmma_gemm_kernel<<<MT, b256, 0, stream>>>(xbf, nullptr, Winbf, b_in, hA,
                                              128, 0, 128, 128, 1);
    cvt_bf16_kernel<<<4096, b256, 0, stream>>>(hA, hbf, (long)NH);

    float* hcur = hA; float* hnext = hB;
    for (int l = 0; l < 2; ++l) {
        zero_kernel<<<8192, b256, 0, stream>>>(agg, (long)NH);
        edge_kernel<<<4096, b256, 0, stream>>>(hcur, ei, agg, E);
        cvt_bf16_kernel<<<4096, b256, 0, stream>>>(agg, aggbf, (long)NH);
        // h' = relu([h | agg] @ Wd[l]^T + bd[l])
        wmma_gemm_kernel<<<MT, b256, 0, stream>>>(hbf, aggbf,
                                                  Wdbf + (size_t)l * 128 * 256,
                                                  bd + l * 128, hnext,
                                                  128, 128, 128, 128, 1);
        cvt_bf16_kernel<<<4096, b256, 0, stream>>>(hnext, hbf, (long)NH);
        // fused tau + GRU + mask, in place on hnext
        gru_kernel<<<MT, b256, 0, stream>>>(hnext, hbf, aggbf, Wihbf, Whhbf,
                                            b_ih, b_hh, Wt, bt, tau_out);
        cvt_bf16_kernel<<<4096, b256, 0, stream>>>(hnext, hbf, (long)NH);
        float* tmp = hcur; hcur = hnext; hnext = tmp;
    }

    // out = h @ Wo^T + bo  (Nout = 2)
    wmma_gemm_kernel<<<MT, b256, 0, stream>>>(hbf, nullptr, Wobf, bo, out,
                                              128, 0, 2, 2, 0);
}
